// LSTM_25546465477223
// MI455X (gfx1250) — compile-verified, matrix operand order checked by reading, not run on hardware
//
#include <hip/hip_runtime.h>

// Problem constants (from reference): T=128, B=256, E=1024, L=1024, V=95
#define T_STEPS 128
#define BATCH   256
#define EDIM    1024
#define LDIM    1024
#define VOCAB   95
#define KDIM    (EDIM + LDIM)      // 2048
#define KC      32                 // K per WMMA
#define KT      (KDIM / KC)        // 64 k-tiles
#define KT_E    (EDIM / KC)        // 32 k-tiles in the embedding phase
#define NT_PG   (LDIM / 16)        // 64 n-tiles per gate
#define WB_ELEMS ((size_t)4 * NT_PG * KT * 32 * 16)   // 8,388,608 bf16 = 16 MiB

typedef __attribute__((ext_vector_type(16))) __bf16 bf16x16;
typedef __attribute__((ext_vector_type(8)))  __bf16 bf16x8;
typedef __attribute__((ext_vector_type(8)))  float  v8f;
typedef __attribute__((ext_vector_type(4)))  unsigned int u32x4;
typedef __attribute__((ext_vector_type(8)))  int          i32x8;
typedef __attribute__((ext_vector_type(4)))  int          i32x4;

// Typed address-space pointers for the async->LDS builtin
// (param 1 is a global (AS1) pointer to v4i32 per compiler diagnostic).
typedef __attribute__((address_space(1))) i32x4* g_v4i_p;
typedef __attribute__((address_space(3))) i32x4* l_v4i_p;

#ifndef __has_builtin
#define __has_builtin(x) 0
#endif

// CDNA5 async global->LDS per-lane copy (ASYNCcnt)
#if __has_builtin(__builtin_amdgcn_global_load_async_to_lds_b128)
#define USE_ASYNC_LDS 1
#else
#define USE_ASYNC_LDS 0
#endif

// CDNA5 Tensor Data Mover (TENSORcnt)
#if __has_builtin(__builtin_amdgcn_tensor_load_to_lds) && \
    __has_builtin(__builtin_amdgcn_s_wait_tensorcnt)
#define USE_TDM 1
#else
#define USE_TDM 0
#endif

__device__ __forceinline__ void wait_async0() {
#if __has_builtin(__builtin_amdgcn_s_wait_asynccnt)
  __builtin_amdgcn_s_wait_asynccnt(0);
#else
  asm volatile("s_wait_asynccnt 0x0" ::: "memory");
#endif
}

__device__ __forceinline__ float fast_sigmoid(float x) {
  return 1.0f / (1.0f + __expf(-x));
}
__device__ __forceinline__ float fast_tanh(float x) {
  float e = __expf(2.0f * x);
  return (e - 1.0f) / (e + 1.0f);
}

#if USE_TDM
// TDM: load a dense [64 rows x 64 bytes] tile of hprev (row stride L*2B = 2048B)
// into LDS at lds_byte_addr, packed row-major (matches At layout).
// D# fields per CDNA5 ISA (data_size=8B units).
__device__ __forceinline__ void tdm_load_h_tile(const __bf16* gsrc,
                                                unsigned lds_byte_addr) {
  unsigned long long ga = (unsigned long long)(size_t)gsrc;
  u32x4 g0;
  g0[0] = 1u;                                              // count=1, user mode
  g0[1] = lds_byte_addr;                                   // lds_addr
  g0[2] = (unsigned)(ga & 0xFFFFFFFFu);                    // global_addr[31:0]
  g0[3] = (unsigned)((ga >> 32) & 0x1FFFFFFu) | (2u << 30);// addr[56:32] | type=2
  i32x8 g1;
  g1[0] = (3 << 16);            // workgroup_mask=0, data_size=3 (8 bytes)
  g1[1] = (int)(256u << 16);    // tensor_dim0 = 256 units (2048B row) [15:0]
  g1[2] = (int)(256u << 16);    // tensor_dim0 hi=0 | tensor_dim1 = 256 rows
  g1[3] = (8 << 16);            // tensor_dim1 hi=0 | tile_dim0 = 8 units (64B)
  g1[4] = 64;                   // tile_dim1 = 64 rows, tile_dim2 = 0
  g1[5] = 256;                  // tensor_dim0_stride = 256 units
  g1[6] = 0;                    // stride0 hi | stride1 lo
  g1[7] = 0;                    // stride1 hi
  i32x4 z4 = {0, 0, 0, 0};
  i32x8 z8 = {0, 0, 0, 0, 0, 0, 0, 0};
  // 6-arg form (clang-23 / therock HIP headers): (g0, g1, g2, g3, g4, cpol)
  __builtin_amdgcn_tensor_load_to_lds(g0, g1, z4, z4, z8, 0);
}
#endif

// Repack the four gate weight matrices (each [L, K] row-major fp32) into a
// bf16 fragment-native layout:
//   WB[(((g*NT_PG + nt)*KT + kt)*32 + lane)*16 + e] = Wg[n, k]
//   n = nt*16 + (lane & 15);  k = kt*32 + (lane >> 4)*16 + e
__global__ __launch_bounds__(256) void pack_weights_kernel(
    const float* __restrict__ Wf, const float* __restrict__ Wi,
    const float* __restrict__ Wo, const float* __restrict__ Wc,
    __bf16* __restrict__ WB) {
  unsigned idx = blockIdx.x * 256u + threadIdx.x;   // < 2^23
  unsigned e    = idx & 15u;
  unsigned lane = (idx >> 4) & 31u;
  unsigned kt   = (idx >> 9) & 63u;
  unsigned nt   = (idx >> 15) & 63u;
  unsigned g    = (idx >> 21) & 3u;
  const float* W = (g == 0) ? Wf : (g == 1) ? Wi : (g == 2) ? Wo : Wc;
  unsigned n = nt * 16u + (lane & 15u);
  unsigned k = kt * KC + (lane >> 4) * 16u + e;
  WB[idx] = (__bf16)W[(size_t)n * KDIM + k];
}

__global__ __launch_bounds__(256) void pack_emb_kernel(
    const float* __restrict__ emb, __bf16* __restrict__ embB, int n) {
  int i = blockIdx.x * 256 + threadIdx.x;
  if (i < n) embB[i] = (__bf16)emb[i];
}

// One recurrent step. Block tile: 64 batch rows x 32 L-cols, all 4 gates.
// 8 waves: (wrow 0..3) x (wcol 0..1); each wave owns a 16x16 output tile and
// runs 4 WMMAs per k-step (one per gate) sharing a single A fragment.
__global__ __launch_bounds__(256) void lstm_step_kernel(
    const int*    __restrict__ xt,     // x[t, :]  (B tokens)
    const __bf16* __restrict__ WB,     // packed weights (L2-resident)
    const __bf16* __restrict__ embB,   // bf16 embedding table
    const __bf16* __restrict__ hprev,  // [B, L] bf16 previous hidden
    const float*  __restrict__ bfb, const float* __restrict__ bib,
    const float*  __restrict__ bob, const float* __restrict__ bcb,
    float*        __restrict__ c_state, // [B, L] fp32 cell state
    __bf16*       __restrict__ hnext,   // [B, L] bf16 next hidden
    float*        __restrict__ out_t)   // d_out + t*B*L (fp32)
{
  __shared__ __align__(32) __bf16 At[2][64 * KC];   // 2 x 4 KiB, double buffered

  const int tid  = threadIdx.x;
  const int lane = tid & 31;
  const int wave = tid >> 5;
  const int wrow = wave >> 1;                 // 0..3
  const int wcol = wave & 1;                  // 0..1
  const int r0   = blockIdx.y * 64;           // batch base
  const int c0   = blockIdx.x * 32;           // L-col base
  const int nt   = blockIdx.x * 2 + wcol;     // n-tile within a gate

  // Hoisted per-thread staging pointers (keeps the k-loop address math flat).
  const int row = tid >> 2;                   // 0..63
  const int seg = (tid & 3) * 8;              // 0,8,16,24 (bf16 elems)
  const int tok = xt[r0 + row];
  const __bf16* emb_src   = embB  + (size_t)tok * EDIM + seg;
  const __bf16* h_row_src = hprev + (size_t)(r0 + row) * LDIM + seg;
  const __bf16* h_tile    = hprev + (size_t)r0 * LDIM;     // TDM tile base
  (void)h_row_src; (void)h_tile;

  // Stage entry[r0..r0+63][kt*KC .. +KC) into At[buf].
  // kt < KT_E: embedding gather (per-lane async->LDS).
  // kt >= KT_E: dense h tile (TDM descriptor, issued by wave 0 only).
  auto stage = [&](int kt, int buf) {
    if (kt < KT_E) {
      const __bf16* src = emb_src + kt * KC;
      __bf16* dst = &At[buf][row * KC + seg];
#if USE_ASYNC_LDS
      __builtin_amdgcn_global_load_async_to_lds_b128(
          (g_v4i_p)(void*)const_cast<__bf16*>(src), (l_v4i_p)(void*)dst, 0, 0);
#else
      *(bf16x8*)dst = *(const bf16x8*)src;
#endif
    } else {
#if USE_TDM
      if (wave == 0) {
        unsigned lds_b = (unsigned)(size_t)(void*)&At[buf][0];
        tdm_load_h_tile(h_tile + (size_t)(kt - KT_E) * KC, lds_b);
      }
#else
      *(bf16x8*)(&At[buf][row * KC + seg]) =
          *(const bf16x8*)(h_row_src + (size_t)(kt - KT_E) * KC);
#endif
    }
  };

  auto stage_wait = [&]() {
#if USE_ASYNC_LDS
    wait_async0();
#endif
#if USE_TDM
    __builtin_amdgcn_s_wait_tensorcnt(0);
#endif
  };

  v8f acc0 = {}, acc1 = {}, acc2 = {}, acc3 = {};

  const size_t gstride = (size_t)NT_PG * KT * 32 * 16;          // per-gate elems
  const __bf16* bptr = WB + (((size_t)nt * KT) * 32 + (size_t)lane) * 16;
  // gate g, k-tile kt fragment at: bptr + g*gstride + kt*512

  stage(0, 0);
  stage_wait();
  __syncthreads();

  for (int kt = 0; kt < KT; ++kt) {
    const int cur = kt & 1;
    if (kt + 1 < KT) stage(kt + 1, cur ^ 1);

    // Keep upcoming weight fragments warm (global_prefetch_b8 path).
    if (kt + 8 < KT) __builtin_prefetch(bptr + (size_t)(kt + 8) * 512, 0, 3);

    bf16x16 a = *(const bf16x16*)(
        &At[cur][(wrow * 16 + (lane & 15)) * KC + (lane >> 4) * 16]);
    bf16x16 b0 = *(const bf16x16*)(bptr + 0 * gstride + (size_t)kt * 512);
    bf16x16 b1 = *(const bf16x16*)(bptr + 1 * gstride + (size_t)kt * 512);
    bf16x16 b2 = *(const bf16x16*)(bptr + 2 * gstride + (size_t)kt * 512);
    bf16x16 b3 = *(const bf16x16*)(bptr + 3 * gstride + (size_t)kt * 512);

    acc0 = __builtin_amdgcn_wmma_f32_16x16x32_bf16(false, a, false, b0,
                                                   (short)0, acc0, false, false);
    acc1 = __builtin_amdgcn_wmma_f32_16x16x32_bf16(false, a, false, b1,
                                                   (short)0, acc1, false, false);
    acc2 = __builtin_amdgcn_wmma_f32_16x16x32_bf16(false, a, false, b2,
                                                   (short)0, acc2, false, false);
    acc3 = __builtin_amdgcn_wmma_f32_16x16x32_bf16(false, a, false, b3,
                                                   (short)0, acc3, false, false);

    stage_wait();      // staged buffer (cur^1) complete before the barrier
    __syncthreads();
  }

  // Fused LSTM cell update. C/D layout: VGPR vi holds M = vi + 8*(lane>=16),
  // N = lane & 15 within the 16x16 tile.
  const int col = c0 + wcol * 16 + (lane & 15);
  const float bfv = bfb[col], biv = bib[col], bov = bob[col], bcv = bcb[col];
  const int rbase = r0 + wrow * 16 + ((lane >> 4) << 3);

#pragma unroll
  for (int vi = 0; vi < 8; ++vi) {
    const int rowg = rbase + vi;
    const size_t idx = (size_t)rowg * LDIM + col;
    float f  = acc0[vi] + bfv;
    float ig = acc1[vi] + biv;
    float o  = acc2[vi] + bov;
    float cg = acc3[vi] + bcv;
    float cp = c_state[idx];
    float cn = fast_sigmoid(f) * cp + fast_sigmoid(ig) * fast_tanh(cg);
    float hn = fast_sigmoid(o) * fast_tanh(cn);
    c_state[idx] = cn;
    hnext[idx]   = (__bf16)hn;
    out_t[idx]   = hn;
  }
}

extern "C" void kernel_launch(void* const* d_in, const int* in_sizes, int n_in,
                              void* d_out, int out_size, void* d_ws, size_t ws_size,
                              hipStream_t stream) {
  (void)in_sizes; (void)n_in; (void)out_size; (void)ws_size;

  // setup_inputs order: x, h, emb, Wf, bf, Wi, bi, Wo, bo, Wc, bc
  const int*   x   = (const int*)  d_in[0];
  const float* emb = (const float*)d_in[2];
  const float* Wf  = (const float*)d_in[3];
  const float* bfb = (const float*)d_in[4];
  const float* Wi  = (const float*)d_in[5];
  const float* bib = (const float*)d_in[6];
  const float* Wo  = (const float*)d_in[7];
  const float* bob = (const float*)d_in[8];
  const float* Wc  = (const float*)d_in[9];
  const float* bcb = (const float*)d_in[10];
  float* out = (float*)d_out;

  // Workspace layout (~18.5 MiB total)
  char* ws = (char*)d_ws;
  __bf16* WB    = (__bf16*)(ws);                                       // 16 MiB
  __bf16* embB  = (__bf16*)(ws + (16u << 20));                          // 190 KiB
  float*  cst   = (float*) (ws + (16u << 20) + (1u << 19));             // 1 MiB
  __bf16* hbuf0 = (__bf16*)(ws + (16u << 20) + (1u << 19) + (1u << 20));// 512 KiB
  __bf16* hbuf1 = (__bf16*)(ws + (16u << 20) + (1u << 19) + (1u << 20) + (1u << 19));

  (void)hipMemsetAsync(cst,   0, (size_t)BATCH * LDIM * sizeof(float),  stream);
  (void)hipMemsetAsync(hbuf0, 0, (size_t)BATCH * LDIM * sizeof(__bf16), stream);

  pack_weights_kernel<<<(unsigned)(WB_ELEMS / 256), 256, 0, stream>>>(
      Wf, Wi, Wo, Wc, WB);
  pack_emb_kernel<<<(VOCAB * EDIM + 255) / 256, 256, 0, stream>>>(
      emb, embB, VOCAB * EDIM);

  for (int t = 0; t < T_STEPS; ++t) {
    __bf16* hp = (t & 1) ? hbuf1 : hbuf0;
    __bf16* hn = (t & 1) ? hbuf0 : hbuf1;
    lstm_step_kernel<<<dim3(LDIM / 32, BATCH / 64), 256, 0, stream>>>(
        x + (size_t)t * BATCH, WB, embB, hp,
        bfb, bib, bob, bcb,
        cst, hn, out + (size_t)t * BATCH * LDIM);
  }
}
